// MoEFFN_7069516169336
// MI455X (gfx1250) — matrix operand order repaired and checked by reading, not compiled, not run on hardware
//
#include <hip/hip_runtime.h>
#include <hip/hip_bf16.h>
#include <math.h>

// ---- problem constants (match reference) ----
#define D_MODEL   1024
#define FFN_DIM   2048
#define NUM_GROUPS 4
#define EPG        2      // experts per group
#define TILE_M    32      // tokens per block tile (2 WMMA M-halves)
#define FC       128      // FFN_DIM chunk per iteration (8 waves x 16 cols)
#define FCP (FC + 8)      // padded ldsH row stride (bank-conflict avoidance)

typedef __bf16 bf16_t;
typedef bf16_t v16bf __attribute__((ext_vector_type(16)));
typedef bf16_t v8bf  __attribute__((ext_vector_type(8)));
typedef float  v8f   __attribute__((ext_vector_type(8)));
typedef float  f32x4 __attribute__((ext_vector_type(4)));
typedef unsigned int u32x4 __attribute__((ext_vector_type(4)));
typedef int    i32x8 __attribute__((ext_vector_type(8)));
typedef int    i32x4 __attribute__((ext_vector_type(4)));

__device__ __forceinline__ bf16_t tobf(float f) { return (bf16_t)f; }

__device__ __forceinline__ v16bf cat8(v8bf lo, v8bf hi) {
    v16bf r;
#pragma unroll
    for (int i = 0; i < 8; ++i) { r[i] = lo[i]; r[8 + i] = hi[i]; }
    return r;
}

// v_rcp_f32-based sigmoid: we are memory-bound, no need for IEEE div chains
__device__ __forceinline__ float sigmoidf_fast(float v) {
    return __builtin_amdgcn_rcpf(1.0f + __expf(-v));
}

// TDM writes 1 pad DWORD after every 256 DWORDs => LDS dword address skew.
// (exact for the 2nd descriptor too: its base, 16*1024 dwords, is 256-aligned)
__device__ __forceinline__ int pad_idx(int idx) { return idx + (idx >> 8); }

// Issue one gather-mode TDM: 16 scattered fp32 rows of x[T][D_MODEL] -> LDS.
__device__ __forceinline__ void tdm_gather16(const float* x, unsigned lds_addr,
                                             const unsigned* tok, int T) {
    unsigned long long ga = (unsigned long long)(const void*)x;
    u32x4 g0;
    g0[0] = 1u | (1u << 31);                  // count=1, gather_mode=1, 16-bit idx
    g0[1] = lds_addr;
    g0[2] = (unsigned)ga;
    g0[3] = (unsigned)((ga >> 32) & 0x01FFFFFFu) | (2u << 30);   // type=2
    i32x8 g1;
    g1[0] = (2 << 16)                         // data_size = 4B
          | (1 << 20)                         // pad_enable
          | (7 << 22)                         // pad_interval: 256 dwords
          | (0 << 25);                        // pad_amount: 1 dword
    g1[1] = (D_MODEL & 0xFFFF) << 16;         // tensor_dim0[15:0]
    g1[2] = ((D_MODEL >> 16) & 0xFFFF) | ((T & 0xFFFF) << 16);
    g1[3] = ((T >> 16) & 0xFFFF) | (D_MODEL << 16);  // dim1 hi | tile_dim0
    g1[4] = 16;                               // tile_dim1 = #gather indices
    g1[5] = D_MODEL;                          // tensor_dim0_stride lo32
    g1[6] = 0;
    g1[7] = 0;
    i32x4 g2, g3;
#pragma unroll
    for (int i = 0; i < 4; ++i) {
        g2[i] = (int)(tok[2 * i]     | (tok[2 * i + 1] << 16));
        g3[i] = (int)(tok[8 + 2 * i] | (tok[8 + 2 * i + 1] << 16));
    }
#if defined(__clang_major__) && (__clang_major__ >= 23)
    i32x8 gz = {};
    __builtin_amdgcn_tensor_load_to_lds(g0, g1, g2, g3, gz, 0);
#else
    __builtin_amdgcn_tensor_load_to_lds(g0, g1, g2, g3, 0);
#endif
}

// ---------------------------------------------------------------------------
// Kernel 0: reset workspace (cnt + perm pad); d_ws is not re-poisoned between
// timed replays, so this must run every launch.
// ---------------------------------------------------------------------------
__global__ void moe_init(int* __restrict__ cnt, int* __restrict__ perm, int permLen) {
    int i = blockIdx.x * blockDim.x + threadIdx.x;
    if (i < 16) cnt[i] = 0;
    if (i < permLen) perm[i] = -1;
}

// ---------------------------------------------------------------------------
// Kernel 1: routing. One 128-thread block per token.
// ---------------------------------------------------------------------------
__global__ __launch_bounds__(128) void moe_route(
    const float* __restrict__ x, const float* __restrict__ macro_w,
    const float* __restrict__ micro_w, int* __restrict__ cnt,
    int* __restrict__ perm, float* __restrict__ wexp, int T)
{
    __shared__ float red[12 * 128];
    const int t   = blockIdx.x;
    const int tid = threadIdx.x;

    float p[12];
#pragma unroll
    for (int s = 0; s < 12; ++s) p[s] = 0.f;

    const float* xr = x + (size_t)t * D_MODEL;
    for (int d = tid; d < D_MODEL; d += 128) {
        float xv = xr[d];
#pragma unroll
        for (int g = 0; g < NUM_GROUPS; ++g) p[g] += xv * macro_w[d * NUM_GROUPS + g];
#pragma unroll
        for (int g = 0; g < NUM_GROUPS; ++g) {
            const float* mw = micro_w + (size_t)g * D_MODEL * EPG + (size_t)d * EPG;
            p[4 + 2 * g + 0] += xv * mw[0];
            p[4 + 2 * g + 1] += xv * mw[1];
        }
    }
#pragma unroll
    for (int s = 0; s < 12; ++s) red[s * 128 + tid] = p[s];
    __syncthreads();
    for (int st = 64; st > 0; st >>= 1) {
        if (tid < st) {
#pragma unroll
            for (int s = 0; s < 12; ++s) red[s * 128 + tid] += red[s * 128 + tid + st];
        }
        __syncthreads();
    }
    if (tid == 0) {
        float ms[NUM_GROUPS];
#pragma unroll
        for (int g = 0; g < NUM_GROUPS; ++g) ms[g] = sigmoidf_fast(red[g * 128]);
        int bg = 0;
#pragma unroll
        for (int g = 1; g < NUM_GROUPS; ++g) if (ms[g] > ms[bg]) bg = g;
        float mv = ms[bg] * __builtin_amdgcn_rcpf(ms[bg] + 1e-9f);
        float s0 = sigmoidf_fast(red[(4 + 2 * bg + 0) * 128]);
        float s1 = sigmoidf_fast(red[(4 + 2 * bg + 1) * 128]);
        float inv = mv * __builtin_amdgcn_rcpf(s0 + s1 + 1e-9f);
        wexp[2 * t + 0] = s0 * inv;
        wexp[2 * t + 1] = s1 * inv;
        int pos = atomicAdd(&cnt[bg], 1);
        perm[bg * T + pos] = t;
    }
}

// ---------------------------------------------------------------------------
// Kernel 2: fused SwiGLU expert FFN, bf16 WMMA, 32 tokens per block.
// X tile staged by two gather-mode TDM descriptors (16 row indices each),
// fp32 into LDS with 1-dword pad per 256 dwords (conflict-free A-fragments).
// Each B-fragment (strided weight gather) feeds both M-halves => 2 WMMAs per
// fragment; 16 persistent f32 accumulators cover all 1024 output dims.
// ---------------------------------------------------------------------------
__global__ __launch_bounds__(256) void moe_expert_ffn(
    const float* __restrict__ x, const float* __restrict__ gate_w,
    const float* __restrict__ up_w, const float* __restrict__ down_w,
    const int* __restrict__ cnt, const int* __restrict__ perm,
    const float* __restrict__ wexp, float* __restrict__ out,
    int T, int tilesPG)
{
    __shared__ alignas(16) float  ldsXf[TILE_M * D_MODEL + 128]; // ~128.5 KB fp32 X (TDM dest, padded)
    __shared__ alignas(16) bf16_t ldsH[TILE_M * FCP];            // ~8.5 KB silu(g)*u chunk
    __shared__ float ldsWr[EPG][TILE_M];
    __shared__ int   ldsTok[TILE_M];

    const int bx   = blockIdx.x;
    const int g    = bx / tilesPG;
    const int tile = bx - g * tilesPG;
    const int tcnt = cnt[g];
    if (tile * TILE_M >= tcnt) return;   // block-uniform early exit

    const int tid  = threadIdx.x;
    const int wave = tid >> 5;
    const int lane = tid & 31;
    const int hi   = lane >> 4;          // wave32 half select
    const int nl   = lane & 15;

    if (tid < TILE_M) {
        int p = perm[g * T + tile * TILE_M + tid];
        ldsTok[tid]   = p;
        ldsWr[0][tid] = (p >= 0) ? wexp[2 * p + 0] : 0.f;
        ldsWr[1][tid] = (p >= 0) ? wexp[2 * p + 1] : 0.f;
    }
    __syncthreads();

    // ---- TDM gathers: 32 token rows of x -> LDS (fp32, padded) ----
    if (wave == 0) {
        unsigned tok[TILE_M];
#pragma unroll
        for (int r = 0; r < TILE_M; ++r) {
            int p = ldsTok[r];
            tok[r] = (unsigned)(p >= 0 ? p : 0);  // pad rows: weight=0 kills them
        }
        unsigned base = (unsigned)(size_t)(void*)ldsXf;
        tdm_gather16(x, base, &tok[0], T);
        tdm_gather16(x, base + (unsigned)((16 * D_MODEL + 64) * 4), &tok[16], T);
        __builtin_amdgcn_s_wait_tensorcnt(0);
    }
    __syncthreads();   // publish TDM-filled LDS to all waves

    v8f zero = {};
    v8f acc[16];       // [tt][m-half]: 8 D-tiles x 2 M-halves
#pragma unroll
    for (int i = 0; i < 16; ++i) acc[i] = zero;

    for (int e = 0; e < EPG; ++e) {
        const float* Wg = gate_w + (size_t)(g * EPG + e) * D_MODEL * FFN_DIM;
        const float* Wu = up_w   + (size_t)(g * EPG + e) * D_MODEL * FFN_DIM;
        const float* Wd = down_w + (size_t)(g * EPG + e) * FFN_DIM * D_MODEL;

        for (int fcb = 0; fcb < FFN_DIM; fcb += FC) {
            // hint: pull this chunk's down-projection slab toward L2 early
            __builtin_prefetch(Wd + (size_t)fcb * D_MODEL + (size_t)lane * 64, 0, 1);

            // ---- gate & up: one 16-wide F-subtile per wave, both M-halves ----
            const int f = fcb + wave * 16 + nl;   // B column for this lane
            v8f accG[2] = {zero, zero};
            v8f accU[2] = {zero, zero};
            for (int kb = 0; kb < D_MODEL; kb += 32) {
                // A fragments (one per M-half): fp32 LDS -> bf16
                v16bf A[2];
#pragma unroll
                for (int th = 0; th < 2; ++th) {
                    int idx0 = (th * 16 + nl) * D_MODEL + kb + hi * 16;
                    int idx1 = idx0 + 8;
                    const f32x4* q0 = (const f32x4*)(ldsXf + pad_idx(idx0));
                    const f32x4* q1 = (const f32x4*)(ldsXf + pad_idx(idx1));
                    f32x4 a0 = q0[0], a1 = q0[1];
                    f32x4 b0 = q1[0], b1 = q1[1];
#pragma unroll
                    for (int i = 0; i < 4; ++i) {
                        A[th][i]      = tobf(a0[i]);
                        A[th][4 + i]  = tobf(a1[i]);
                        A[th][8 + i]  = tobf(b0[i]);
                        A[th][12 + i] = tobf(b1[i]);
                    }
                }

                v16bf Bg, Bu;
                const float* wg0 = Wg + (size_t)(kb + hi * 16) * FFN_DIM + f;
                const float* wu0 = Wu + (size_t)(kb + hi * 16) * FFN_DIM + f;
#pragma unroll
                for (int j = 0; j < 16; ++j) {
                    Bg[j] = tobf(wg0[(size_t)j * FFN_DIM]);
                    Bu[j] = tobf(wu0[(size_t)j * FFN_DIM]);
                }
                // one B-fragment feeds both M-halves
                accG[0] = __builtin_amdgcn_wmma_f32_16x16x32_bf16(
                    false, A[0], false, Bg, (short)0, accG[0], false, false);
                accG[1] = __builtin_amdgcn_wmma_f32_16x16x32_bf16(
                    false, A[1], false, Bg, (short)0, accG[1], false, false);
                accU[0] = __builtin_amdgcn_wmma_f32_16x16x32_bf16(
                    false, A[0], false, Bu, (short)0, accU[0], false, false);
                accU[1] = __builtin_amdgcn_wmma_f32_16x16x32_bf16(
                    false, A[1], false, Bu, (short)0, accU[1], false, false);
            }

            __syncthreads();  // previous chunk's ldsH consumers done
            // H = route_weight[token] * silu(G) * U   (bf16 to LDS)
#pragma unroll
            for (int th = 0; th < 2; ++th) {
#pragma unroll
                for (int r = 0; r < 8; ++r) {
                    int m = th * 16 + r + 8 * hi;    // C/D layout: VGPR r, half hi
                    float gv = accG[th][r];
                    float uv = accU[th][r];
                    float h  = ldsWr[e][m] * (gv * sigmoidf_fast(gv)) * uv;
                    ldsH[m * FCP + wave * 16 + nl] = tobf(h);
                }
            }
            __syncthreads();

            // ---- down projection: out[32 x 1024] += H[32 x FC] @ Wd[FC x 1024]
#pragma unroll
            for (int tt = 0; tt < 8; ++tt) {
                const int dn = wave * 128 + tt * 16 + nl;  // B column for this lane
#pragma unroll
                for (int ks = 0; ks < FC; ks += 32) {
                    v16bf Bd;
                    const float* wd0 = Wd + (size_t)(fcb + ks + hi * 16) * D_MODEL + dn;
#pragma unroll
                    for (int j = 0; j < 16; ++j)
                        Bd[j] = tobf(wd0[(size_t)j * D_MODEL]);

#pragma unroll
                    for (int th = 0; th < 2; ++th) {
                        const v8bf* ph0 = (const v8bf*)(ldsH + (th * 16 + nl) * FCP + ks + hi * 16);
                        const v8bf* ph1 = (const v8bf*)(ldsH + (th * 16 + nl) * FCP + ks + hi * 16 + 8);
                        v16bf Ah = cat8(*ph0, *ph1);
                        acc[tt * 2 + th] = __builtin_amdgcn_wmma_f32_16x16x32_bf16(
                            false, Ah, false, Bd, (short)0, acc[tt * 2 + th], false, false);
                    }
                }
            }
        }
    }

    __syncthreads();
    // store: each token row written exactly once across the whole grid
#pragma unroll
    for (int tt = 0; tt < 8; ++tt) {
        const int dn = wave * 128 + tt * 16 + nl;
#pragma unroll
        for (int th = 0; th < 2; ++th) {
#pragma unroll
            for (int r = 0; r < 8; ++r) {
                int m = th * 16 + r + 8 * hi;
                int p = ldsTok[m];
                if (p >= 0) out[(size_t)p * D_MODEL + dn] = acc[tt * 2 + th][r];
            }
        }
    }
}

// ---------------------------------------------------------------------------
extern "C" void kernel_launch(void* const* d_in, const int* in_sizes, int n_in,
                              void* d_out, int out_size, void* d_ws, size_t ws_size,
                              hipStream_t stream) {
    const float* x       = (const float*)d_in[0];
    const float* macro_w = (const float*)d_in[1];
    const float* micro_w = (const float*)d_in[2];
    const float* gate_w  = (const float*)d_in[3];
    const float* up_w    = (const float*)d_in[4];
    const float* down_w  = (const float*)d_in[5];
    float* out = (float*)d_out;

    const int T = in_sizes[0] / D_MODEL;   // tokens (B*S)

    // workspace layout: cnt[16] | perm[4*T] | wexp[2*T]
    int*   cnt  = (int*)d_ws;
    int*   perm = cnt + 16;
    float* wexp = (float*)(perm + NUM_GROUPS * T);

    const int permLen = NUM_GROUPS * T;
    moe_init<<<(permLen + 255) / 256, 256, 0, stream>>>(cnt, perm, permLen);
    moe_route<<<T, 128, 0, stream>>>(x, macro_w, micro_w, cnt, perm, wexp, T);

    const int tilesPG = (T + TILE_M - 1) / TILE_M;
    moe_expert_ffn<<<NUM_GROUPS * tilesPG, 256, 0, stream>>>(
        x, gate_w, up_w, down_w, cnt, perm, wexp, out, T, tilesPG);
}
